// EMAVectorQuantizer_55121610277368
// MI455X (gfx1250) — matrix operand order repaired and checked by reading, not compiled, not run on hardware
//
#include <hip/hip_runtime.h>
#include <hip/hip_bf16.h>
#include <stdint.h>

// ---------------- problem constants (from reference) ----------------
#define K_CODES 1024
#define DDIM    128
#define BB      8
#define LL      4096
#define NN      (BB*LL)          // 32768 flat latent vectors
#define DECAY_F 0.99f
#define OMD_F   0.01f            // 1 - decay
#define BETA_F  0.25f
#define EPS_F   1e-5f
#define RESTART_F 0.5f

// ---------------- CDNA5 WMMA vector types ----------------
typedef __attribute__((ext_vector_type(16))) __bf16 v16bf;
typedef __attribute__((ext_vector_type(8)))  __bf16 v8bf;
typedef __attribute__((ext_vector_type(8)))  float  v8f;

// =====================================================================
// Kernel 1: e_sq, f32->bf16 codebook, zero accumulators
// grid: K_CODES blocks x DDIM threads
// =====================================================================
__global__ void vq_prep(const float* __restrict__ emb,
                        __bf16* __restrict__ ebf, float* __restrict__ esq,
                        float* __restrict__ counts, float* __restrict__ wsum,
                        float* __restrict__ scalars /* [0]=mseSum [1]=nTotal */) {
  const int k = blockIdx.x, d = threadIdx.x;
  float v = emb[k * DDIM + d];
  ebf[k * DDIM + d] = (__bf16)v;
  wsum[(size_t)k * DDIM + d] = 0.f;
  __shared__ float red[DDIM];
  red[d] = v * v;
  __syncthreads();
  for (int s = DDIM / 2; s > 0; s >>= 1) {
    if (d < s) red[d] += red[d + s];
    __syncthreads();
  }
  if (d == 0) { esq[k] = red[0]; counts[k] = 0.f; }
  if (k == 0 && d == 0) { scalars[0] = 0.f; scalars[1] = 0.f; }
}

// =====================================================================
// Kernel 2: nearest-code search via bf16 WMMA.
// Block = 256 threads (8 wave32s), 128 z-rows per block.
// Each wave owns a fixed 16-row M tile: its 4 A fragments (D=128) are
// loaded from LDS ONCE into registers; the wave then sweeps all K codes
// in 64-code LDS chunks (4 n-tiles x 4 kk WMMAs per chunk).
// Argmin is register-resident: C layout pins each lane to one column and
// 8 rows, so each lane carries 8 running (bestV,bestI) pairs; final
// per-row reduce = shfl_xor butterfly over the 16-lane halves.
// score = e_sq - 2*dot  (row-constant z^2 dropped: argmin-invariant)
// =====================================================================
#define MB  128    // z rows per block (8 waves x 16 rows)
#define NB  64     // codes per LDS chunk
#define LDA 136    // padded bf16 row stride (272B -> 16B aligned frags)

__global__ __launch_bounds__(256)
void vq_argmin(const float* __restrict__ z, const __bf16* __restrict__ ebf,
               const float* __restrict__ esq, int* __restrict__ idxInt,
               float* __restrict__ idxOut, float* __restrict__ counts) {
  __shared__ __bf16 As[MB][LDA];
  __shared__ __bf16 Bs[NB][LDA];
  __shared__ float  eSqS[NB];

  const int t    = threadIdx.x;
  const int lane = t & 31;          // wave32
  const int wave = t >> 5;          // 0..7
  const int rowStart = blockIdx.x * MB;      // flat n (never crosses a batch)
  const int b  = rowStart / LL;
  const int l0 = rowStart % LL;

  // ---- stage A tile: z_e[b][d][l0+r], coalesced along l, bf16 into LDS ----
  {
    const int r = t & 127;
    const int dbase = t >> 7;                // 0..1
    #pragma unroll
    for (int it = 0; it < 64; ++it) {
      const int d = it * 2 + dbase;
      As[r][d] = (__bf16)z[((size_t)b * DDIM + d) * LL + l0 + r];
    }
  }
  __syncthreads();

  const int m0  = wave * 16;        // this wave's row tile
  const int r15 = lane & 15;
  const int hi8 = lane >> 4;        // ISA 16-bit A/B layout: lane halves

  // ---- preload the wave's 4 A fragments (whole D=128) into registers ----
  v16bf afrag[4];
  #pragma unroll
  for (int kki = 0; kki < 4; ++kki) {
    const int kk = kki * 32;
    const __bf16* p = &As[m0 + r15][0];
    v8bf lo = *(const v8bf*)(p + kk + hi8 * 8);
    v8bf hv = *(const v8bf*)(p + kk + 16 + hi8 * 8);
    #pragma unroll
    for (int i = 0; i < 8; ++i) { afrag[kki][i] = lo[i]; afrag[kki][i + 8] = hv[i]; }
  }

  float bestV[8];
  int   bestI[8];
  #pragma unroll
  for (int j = 0; j < 8; ++j) { bestV[j] = 3.4e38f; bestI[j] = 0; }

  for (int nc = 0; nc < K_CODES / NB; ++nc) {
    // ---- stage B chunk: 64 codes x 128 bf16; 16B per thread x 4 passes ----
    #pragma unroll
    for (int p4 = 0; p4 < 4; ++p4) {
      const int slot = p4 * 256 + t;         // 0..1023 16B-chunks
      const int code = slot >> 4;
      const int ch   = slot & 15;            // 16 x 8-half chunks per row
      *(v8bf*)&Bs[code][ch * 8] =
          *(const v8bf*)(ebf + (size_t)(nc * NB + code) * DDIM + ch * 8);
    }
    if (t < NB) eSqS[t] = esq[nc * NB + t];
    __syncthreads();

    #pragma unroll
    for (int nt = 0; nt < 4; ++nt) {         // four 16-code tiles per chunk
      v8f acc = {0.f, 0.f, 0.f, 0.f, 0.f, 0.f, 0.f, 0.f};
      #pragma unroll
      for (int kki = 0; kki < 4; ++kki) {
        const int kk = kki * 32;
        v16bf bf;                            // column (nt*16 + r15) of B
        const __bf16* p = &Bs[nt * 16 + r15][0];
        v8bf lo = *(const v8bf*)(p + kk + hi8 * 8);
        v8bf hv = *(const v8bf*)(p + kk + 16 + hi8 * 8);
        #pragma unroll
        for (int i = 0; i < 8; ++i) { bf[i] = lo[i]; bf[i + 8] = hv[i]; }
        acc = __builtin_amdgcn_wmma_f32_16x16x32_bf16(false, afrag[kki], false, bf,
                                                      (short)0, acc, false, false);
      }
      // C layout: vgpr j -> row m0+j (+8 for upper lanes), col fixed per lane
      const int colLoc = nt * 16 + r15;
      const int colGlb = nc * NB + colLoc;
      const float eq = eSqS[colLoc];
      #pragma unroll
      for (int j = 0; j < 8; ++j) {
        const float s = __builtin_fmaf(-2.f, acc[j], eq);
        if (s < bestV[j]) { bestV[j] = s; bestI[j] = colGlb; }
      }
    }
    __syncthreads();   // protect Bs reuse next chunk
  }

  // ---- per-row reduce across the 16-lane halves (shfl_xor butterfly) ----
  #pragma unroll
  for (int j = 0; j < 8; ++j) {
    float v = bestV[j]; int ii = bestI[j];
    #pragma unroll
    for (int m = 1; m < 16; m <<= 1) {
      const float ov = __shfl_xor(v, m, 32);
      const int   oi = __shfl_xor(ii, m, 32);
      if (ov < v || (ov == v && oi < ii)) { v = ov; ii = oi; }
    }
    bestV[j] = v; bestI[j] = ii;
  }
  if (r15 == 0) {               // lanes 0 and 16 hold rows m0..m0+7 / +8..15
    const int rowBase = rowStart + m0 + hi8 * 8;
    #pragma unroll
    for (int j = 0; j < 8; ++j) {
      const int n = rowBase + j;
      idxInt[n] = bestI[j];
      idxOut[n] = (float)bestI[j];
      atomicAdd(&counts[bestI[j]], 1.f);
    }
  }
}

// =====================================================================
// Kernel 3: segment-sum of z rows into wsum[K][D] (f32 global atomics)
// grid: B*D blocks x 256 threads, loop over L (coalesced reads)
// =====================================================================
__global__ void vq_scatter(const float* __restrict__ z,
                           const int* __restrict__ idxInt,
                           float* __restrict__ wsum) {
  const int bd = blockIdx.x;
  const int d = bd & (DDIM - 1);
  const int b = bd >> 7;
  for (int l = threadIdx.x; l < LL; l += blockDim.x) {
    const float v = z[((size_t)b * DDIM + d) * LL + l];
    const int k = idxInt[b * LL + l];
    atomicAdd(&wsum[(size_t)k * DDIM + d], v);
  }
}

// =====================================================================
// Kernel 4a: EMA cluster sizes + dead flags + n = sum(new_cluster)
// single block of 1024 threads (one per code)
// =====================================================================
__global__ void vq_cluster(const float* __restrict__ clusterSize,
                           const float* __restrict__ counts,
                           float* __restrict__ newCluster,
                           float* __restrict__ deadFlag,
                           float* __restrict__ scalars) {
  __shared__ float red[K_CODES];
  const int k = threadIdx.x;
  float c = DECAY_F * clusterSize[k] + OMD_F * counts[k];
  const float dead = (c < RESTART_F) ? 1.f : 0.f;
  if (dead > 0.f) c = RESTART_F;
  newCluster[k] = c;
  deadFlag[k] = dead;
  red[k] = c;
  __syncthreads();
  for (int s = K_CODES / 2; s > 0; s >>= 1) {
    if (k < s) red[k] += red[k + s];
    __syncthreads();
  }
  if (k == 0) scalars[1] = red[0];
}

// =====================================================================
// Kernel 4b: Laplace-smoothed normalization -> updated codebook
// (dead codes restart from a deterministic pseudo-random z row)
// =====================================================================
__device__ inline uint32_t hash_u32(uint32_t x) {
  x ^= x >> 16; x *= 0x7feb352du; x ^= x >> 15; x *= 0x846ca68bu; x ^= x >> 16;
  return x;
}

__global__ void vq_newemb(const float* __restrict__ z,
                          const float* __restrict__ embAvg,
                          const float* __restrict__ wsum,
                          const float* __restrict__ newCluster,
                          const float* __restrict__ deadFlag,
                          const float* __restrict__ scalars,
                          float* __restrict__ newEmb) {
  const int k = blockIdx.x, d = threadIdx.x;
  const float nT = scalars[1];
  const float smoothed = (newCluster[k] + EPS_F) / (nT + K_CODES * EPS_F) * nT;
  float avg;
  if (deadFlag[k] > 0.f) {
    // deterministic stand-in for jax.random.randint(key(42)) restart indices
    const int r = (int)(hash_u32((uint32_t)k ^ 0x9e3779b9u) % NN);
    const int b = r / LL, l = r % LL;
    avg = z[((size_t)b * DDIM + d) * LL + l] * RESTART_F;
  } else {
    avg = DECAY_F * embAvg[(size_t)k * DDIM + d] + OMD_F * wsum[(size_t)k * DDIM + d];
  }
  newEmb[(size_t)k * DDIM + d] = avg / smoothed;
}

// =====================================================================
// Kernel 5: z_q gather with UPDATED codebook (torch aliasing semantics),
// transposed store to (B,D,L), MSE partial sums.
// Forward value of z_q_st == z_q; both losses share the same MSE.
// =====================================================================
__global__ void vq_output(const float* __restrict__ z,
                          const int* __restrict__ idxInt,
                          const float* __restrict__ newEmb,
                          float* __restrict__ outZ,
                          float* __restrict__ scalars) {
  __shared__ float red[256];
  const int bd = blockIdx.x;
  const int d = bd & (DDIM - 1);
  const int b = bd >> 7;
  float acc = 0.f;
  for (int l = threadIdx.x; l < LL; l += blockDim.x) {
    const size_t zi = ((size_t)b * DDIM + d) * LL + l;
    const int k = idxInt[b * LL + l];
    const float q = newEmb[(size_t)k * DDIM + d];
    outZ[zi] = q;
    const float diff = z[zi] - q;
    acc += diff * diff;
  }
  red[threadIdx.x] = acc;
  __syncthreads();
  for (int s = 128; s > 0; s >>= 1) {
    if ((int)threadIdx.x < s) red[threadIdx.x] += red[threadIdx.x + s];
    __syncthreads();
  }
  if (threadIdx.x == 0) atomicAdd(&scalars[0], red[0]);
}

// =====================================================================
// Kernel 6: final loss scalars
// =====================================================================
__global__ void vq_loss(const float* __restrict__ scalars,
                        float* __restrict__ outScalars) {
  const float mse = scalars[0] / (float)((size_t)BB * DDIM * LL);
  outScalars[0] = (1.f + BETA_F) * mse;   // vq_loss
  outScalars[1] = mse;                    // codebook_loss
  outScalars[2] = BETA_F * mse;           // beta * commitment_loss
}

// =====================================================================
extern "C" void kernel_launch(void* const* d_in, const int* in_sizes, int n_in,
                              void* d_out, int out_size, void* d_ws, size_t ws_size,
                              hipStream_t stream) {
  (void)in_sizes; (void)n_in; (void)out_size; (void)ws_size;
  const float* z       = (const float*)d_in[0];   // (B, D, L)
  const float* emb     = (const float*)d_in[1];   // (K, D)
  const float* cluster = (const float*)d_in[2];   // (K,)
  const float* embAvg  = (const float*)d_in[3];   // (K, D)

  char* ws = (char*)d_ws;
  size_t off = 0;
  auto alloc = [&](size_t bytes) {
    void* p = ws + off;
    off = (off + bytes + 255) & ~(size_t)255;
    return p;
  };
  __bf16* ebf     = (__bf16*)alloc((size_t)K_CODES * DDIM * 2);
  float*  esq     = (float*) alloc((size_t)K_CODES * 4);
  float*  counts  = (float*) alloc((size_t)K_CODES * 4);
  float*  wsum    = (float*) alloc((size_t)K_CODES * DDIM * 4);
  int*    idxInt  = (int*)   alloc((size_t)NN * 4);
  float*  newClu  = (float*) alloc((size_t)K_CODES * 4);
  float*  dead    = (float*) alloc((size_t)K_CODES * 4);
  float*  newEmb  = (float*) alloc((size_t)K_CODES * DDIM * 4);
  float*  scalars = (float*) alloc(256);

  float* outZ    = (float*)d_out;                 // B*D*L
  float* outIdx  = outZ + (size_t)BB * DDIM * LL; // N (indices as float)
  float* outLoss = outIdx + NN;                   // 3 scalars

  vq_prep   <<<K_CODES, DDIM, 0, stream>>>(emb, ebf, esq, counts, wsum, scalars);
  vq_argmin <<<NN / MB, 256, 0, stream>>>(z, ebf, esq, idxInt, outIdx, counts);
  vq_scatter<<<BB * DDIM, 256, 0, stream>>>(z, idxInt, wsum);
  vq_cluster<<<1, K_CODES, 0, stream>>>(cluster, counts, newClu, dead, scalars);
  vq_newemb <<<K_CODES, DDIM, 0, stream>>>(z, embAvg, wsum, newClu, dead, scalars, newEmb);
  vq_output <<<BB * DDIM, 256, 0, stream>>>(z, idxInt, newEmb, outZ, scalars);
  vq_loss   <<<1, 1, 0, stream>>>(scalars, outLoss);
}
